// CCIPBatchMetrics_30339648979154
// MI455X (gfx1250) — matrix operand order repaired and checked by reading, not compiled, not run on hardware
//
#include <hip/hip_runtime.h>

// ---------------------------------------------------------------------------
// Types for CDNA5 WMMA (wave32): V_WMMA_F32_16X16X32_BF16
// ---------------------------------------------------------------------------
typedef __attribute__((ext_vector_type(16))) __bf16 v16bf;
typedef __attribute__((ext_vector_type(8)))  __bf16 v8bf;
typedef __attribute__((ext_vector_type(8)))  float  v8f;
typedef __attribute__((ext_vector_type(4)))  unsigned int v4u;
typedef __attribute__((ext_vector_type(8)))  int    v8i;
typedef __attribute__((ext_vector_type(4)))  int    v4i;

#define BDIM 4096
#define DDIM 768
#define EPS  1e-8f

#define KSTEP    32                 // K elements per tile step
#define TILE_ROWS 64                // rows per TDM tile
#define ROW_BYTES 64                // 32 bf16 = 64 B of payload per row
#define PAD_BYTES 16                // TDM pad -> 80 B LDS pitch (bank-conflict free)
#define ROW_PITCH (ROW_BYTES + PAD_BYTES)
#define TILE_BYTES (TILE_ROWS * ROW_PITCH)   // 5120 B

#if defined(__has_builtin)
#if __has_builtin(__builtin_amdgcn_tensor_load_to_lds)
#define HAVE_TDM 1
#endif
#endif

// ---------------------------------------------------------------------------
// bf16 helpers (round-to-nearest-even), header-independent
// ---------------------------------------------------------------------------
static __device__ __forceinline__ unsigned short f2bf_rne(float f) {
    unsigned int u = __float_as_uint(f);
    unsigned int lsb = (u >> 16) & 1u;
    u += 0x7fffu + lsb;
    return (unsigned short)(u >> 16);
}
static __device__ __forceinline__ float bf2f(unsigned short h) {
    return __uint_as_float(((unsigned int)h) << 16);
}

// ---------------------------------------------------------------------------
// Pass 1: row norms + bf16 hi/lo split of normalized rows
// ---------------------------------------------------------------------------
__global__ __launch_bounds__(256) void norm_split_kernel(
    const float* __restrict__ x,
    unsigned short* __restrict__ xh,
    unsigned short* __restrict__ xl)
{
    const int row = blockIdx.x;
    const int tid = threadIdx.x;
    const float* xr = x + (size_t)row * DDIM;

    float ss = 0.0f;
    #pragma unroll
    for (int i = tid; i < DDIM; i += 256) {
        float v = xr[i];
        ss = fmaf(v, v, ss);
    }

    __shared__ float sred[256];
    sred[tid] = ss;
    __syncthreads();
    #pragma unroll
    for (int s = 128; s > 0; s >>= 1) {
        if (tid < s) sred[tid] += sred[tid + s];
        __syncthreads();
    }
    const float inv = 1.0f / fmaxf(sqrtf(sred[0]), EPS);

    unsigned short* hr = xh + (size_t)row * DDIM;
    unsigned short* lr = xl + (size_t)row * DDIM;
    #pragma unroll
    for (int i = tid; i < DDIM; i += 256) {
        float v = xr[i] * inv;
        unsigned short h = f2bf_rne(v);
        unsigned short l = f2bf_rne(v - bf2f(h));
        hr[i] = h;
        lr[i] = l;
    }
}

// ---------------------------------------------------------------------------
// TDM: load a 64-row x 32-elem bf16 tile (row stride 768 elems) into LDS with
// 16 B padding after every 64 B row (CDNA5 ISA descriptor).
//
// D# group0: [1:0]=count=1 | [63:32]=lds_addr | [120:64]=global_addr |
//            [127:126]=type=2
// D# group1: [17:16]=data_size=1(2B) | [20]=pad_enable |
//            [24:22]=pad_interval=3(16 dw) | [31:25]=pad_amount=3(4 dw)
//            [79:48]=tensor_dim0=768 | [111:80]=tensor_dim1=4096
//            [127:112]=tile_dim0=32 | [143:128]=tile_dim1=64
//            [207:160]=tensor_dim0_stride=768
// ---------------------------------------------------------------------------
#if defined(HAVE_TDM)
static __device__ __forceinline__ void tdm_load_tile(
    unsigned int lds_addr, const unsigned short* __restrict__ gbase,
    int row0, int k)
{
    unsigned long long ga =
        (unsigned long long)(const void*)(gbase + (size_t)row0 * DDIM + k);
    v4u g0;
    g0[0] = 1u;                                        // count = 1
    g0[1] = lds_addr;                                  // LDS byte address
    g0[2] = (unsigned int)(ga & 0xffffffffull);        // global_addr[31:0]
    g0[3] = (unsigned int)((ga >> 32) & 0x01ffffffull) // global_addr[56:32]
          | 0x80000000u;                               // type = 2 ("image")
    v8i g1;
    g1[0] = 0x06D10000;   // data_size=2B, pad_en, pad_interval=16dw, pad=4dw
    g1[1] = 0x03000000;   // tensor_dim0 = 768  (bits 79:48, low half)
    g1[2] = 0x10000000;   // tensor_dim1 = 4096 (bits 111:80)
    g1[3] = 0x00200000;   // tile_dim0 = 32     (bits 127:112)
    g1[4] = TILE_ROWS;    // tile_dim1 = 64     (bits 143:128)
    g1[5] = DDIM;         // tensor_dim0_stride = 768 (bits 207:160, low)
    g1[6] = 0;
    g1[7] = 0;
    v4i z4 = {0, 0, 0, 0};
#if __clang_major__ >= 23
    v8i z8 = {0, 0, 0, 0, 0, 0, 0, 0};
    __builtin_amdgcn_tensor_load_to_lds(g0, g1, z4, z4, z8, 0);
#else
    __builtin_amdgcn_tensor_load_to_lds(g0, g1, z4, z4, 0);
#endif
}
#endif

// ---------------------------------------------------------------------------
// Fragment readers (LDS tile, 80 B row pitch) for V_WMMA_F32_16X16X32_BF16.
// A 16x32 (ISA §7.12.2): lane m=lane&15, half=lane>>4;
//   V0..V3: K = half*8+{0..7}; V4..V7: K = 16+half*8+{0..7}
// B 32x16: lane n=lane&15; same K order as A: K = half*8+{0..7}, then 16+half*8+{0..7}.
// ---------------------------------------------------------------------------
static __device__ __forceinline__ v16bf frag_A_lds(
    const unsigned char* tile, int rb, int lane)
{
    const int m = lane & 15, half = lane >> 4;
    const unsigned char* p = tile + (rb + m) * ROW_PITCH;
    union { v16bf v; v8bf h[2]; } u;
    u.h[0] = *reinterpret_cast<const v8bf*>(p + half * 16);
    u.h[1] = *reinterpret_cast<const v8bf*>(p + 32 + half * 16);
    return u.v;
}
static __device__ __forceinline__ v16bf frag_B_lds(
    const unsigned char* tile, int cb, int lane)
{
    const int n = lane & 15, kb = (lane >> 4) * 16;   // byte offset of K group (half*8 elems)
    const unsigned char* p = tile + (cb + n) * ROW_PITCH + kb;
    union { v16bf v; v8bf h[2]; } u;
    u.h[0] = *reinterpret_cast<const v8bf*>(p);
    u.h[1] = *reinterpret_cast<const v8bf*>(p + 32);
    return u.v;
}

static __device__ __forceinline__ v8f wmma_bf16(v16bf a, v16bf b, v8f c) {
    return __builtin_amdgcn_wmma_f32_16x16x32_bf16(
        false, a, false, b, (short)0, c, false, false);
}

// Fallback direct-global fragment loaders (used only if TDM builtin absent)
static __device__ __forceinline__ v16bf load_frag_A(
    const unsigned short* __restrict__ base, int row, int k, int lane)
{
    const int m = lane & 15, half = lane >> 4;
    const __bf16* p =
        reinterpret_cast<const __bf16*>(base + (size_t)(row + m) * DDIM + k);
    union { v16bf v; v8bf h[2]; } u;
    u.h[0] = *reinterpret_cast<const v8bf*>(p + half * 8);
    u.h[1] = *reinterpret_cast<const v8bf*>(p + 16 + half * 8);
    return u.v;
}
static __device__ __forceinline__ v16bf load_frag_B(
    const unsigned short* __restrict__ base, int col, int k, int lane)
{
    const int n = lane & 15, kh = (lane >> 4) * 8;
    const __bf16* p =
        reinterpret_cast<const __bf16*>(base + (size_t)(col + n) * DDIM + k + kh);
    union { v16bf v; v8bf h[2]; } u;
    u.h[0] = *reinterpret_cast<const v8bf*>(p);
    u.h[1] = *reinterpret_cast<const v8bf*>(p + 16);
    return u.v;
}

// ---------------------------------------------------------------------------
// Pass 2: sim tile GEMM (bf16x3 split) + fused Linear(1,2) epilogue.
// 128 threads = 4 waves; WG tile 64x64; wave tile 32x32 (2x2 frags).
// TDM double-buffers four 64x32 tiles (A-hi, A-lo, B-hi, B-lo) in LDS.
// ---------------------------------------------------------------------------
__global__ __launch_bounds__(128) void gemm_bf16x3_kernel(
    const unsigned short* __restrict__ xh,
    const unsigned short* __restrict__ xl,
    const float* __restrict__ fc_w,
    const float* __restrict__ fc_b,
    float* __restrict__ out)
{
    const int lane = threadIdx.x & 31;
    const int wave = threadIdx.x >> 5;
    const int tileR = blockIdx.y * 64;
    const int tileC = blockIdx.x * 64;
    const int waveR = (wave >> 1) * 32;   // wave sub-tile within 64x64
    const int waveC = (wave & 1) * 32;

    v8f acc[2][2] = {};

#if defined(HAVE_TDM)
    __shared__ __align__(16) unsigned char smem[2][4][TILE_BYTES];

    // Issue all four tile loads for K-step `ks` into buffer `buf`.
    auto issue = [&](int buf, int k) {
        tdm_load_tile((unsigned int)(unsigned long long)(void*)smem[buf][0],
                      xh, tileR, k);
        tdm_load_tile((unsigned int)(unsigned long long)(void*)smem[buf][1],
                      xl, tileR, k);
        tdm_load_tile((unsigned int)(unsigned long long)(void*)smem[buf][2],
                      xh, tileC, k);
        tdm_load_tile((unsigned int)(unsigned long long)(void*)smem[buf][3],
                      xl, tileC, k);
    };

    if (wave == 0) issue(0, 0);

    const int NSTEP = DDIM / KSTEP;   // 24
    for (int ks = 0; ks < NSTEP; ++ks) {
        if (wave == 0) __builtin_amdgcn_s_wait_tensorcnt((short)0);
        __syncthreads();              // tiles for step ks visible to all waves
        if (wave == 0 && (ks + 1) < NSTEP) issue((ks + 1) & 1, (ks + 1) * KSTEP);

        const unsigned char* Ah = smem[ks & 1][0];
        const unsigned char* Al = smem[ks & 1][1];
        const unsigned char* Bh = smem[ks & 1][2];
        const unsigned char* Bl = smem[ks & 1][3];

        v16bf ah[2], al[2], bh[2], bl[2];
        #pragma unroll
        for (int i = 0; i < 2; ++i) {
            ah[i] = frag_A_lds(Ah, waveR + 16 * i, lane);
            al[i] = frag_A_lds(Al, waveR + 16 * i, lane);
            bh[i] = frag_B_lds(Bh, waveC + 16 * i, lane);
            bl[i] = frag_B_lds(Bl, waveC + 16 * i, lane);
        }
        #pragma unroll
        for (int mi = 0; mi < 2; ++mi) {
            #pragma unroll
            for (int nj = 0; nj < 2; ++nj) {
                acc[mi][nj] = wmma_bf16(ah[mi], bl[nj], acc[mi][nj]);
                acc[mi][nj] = wmma_bf16(al[mi], bh[nj], acc[mi][nj]);
                acc[mi][nj] = wmma_bf16(ah[mi], bh[nj], acc[mi][nj]);
            }
        }
        __syncthreads();              // done reading buffer ks&1
    }
#else
    for (int k = 0; k < DDIM; k += KSTEP) {
        v16bf ah[2], al[2], bh[2], bl[2];
        #pragma unroll
        for (int i = 0; i < 2; ++i) {
            ah[i] = load_frag_A(xh, tileR + waveR + 16 * i, k, lane);
            al[i] = load_frag_A(xl, tileR + waveR + 16 * i, k, lane);
            bh[i] = load_frag_B(xh, tileC + waveC + 16 * i, k, lane);
            bl[i] = load_frag_B(xl, tileC + waveC + 16 * i, k, lane);
        }
        #pragma unroll
        for (int mi = 0; mi < 2; ++mi) {
            #pragma unroll
            for (int nj = 0; nj < 2; ++nj) {
                acc[mi][nj] = wmma_bf16(ah[mi], bl[nj], acc[mi][nj]);
                acc[mi][nj] = wmma_bf16(al[mi], bh[nj], acc[mi][nj]);
                acc[mi][nj] = wmma_bf16(ah[mi], bh[nj], acc[mi][nj]);
            }
        }
    }
#endif

    // Fused epilogue: out[i,j,c] = sim*w[c] + b[c]
    const float w0 = fc_w[0], w1 = fc_w[1];
    const float b0 = fc_b[0], b1 = fc_b[1];
    const int half = lane >> 4;
    const int nlo  = lane & 15;

    #pragma unroll
    for (int mi = 0; mi < 2; ++mi) {
        #pragma unroll
        for (int nj = 0; nj < 2; ++nj) {
            const int Nc = tileC + waveC + nj * 16 + nlo;
            #pragma unroll
            for (int d = 0; d < 8; ++d) {
                const int Mr = tileR + waveR + mi * 16 + d + 8 * half;
                const float s = acc[mi][nj][d];
                float2 o;
                o.x = fmaf(s, w0, b0);
                o.y = fmaf(s, w1, b1);
                *reinterpret_cast<float2*>(
                    out + ((size_t)Mr * BDIM + Nc) * 2) = o;
            }
        }
    }
}

// ---------------------------------------------------------------------------
// Launcher
// ---------------------------------------------------------------------------
extern "C" void kernel_launch(void* const* d_in, const int* in_sizes, int n_in,
                              void* d_out, int out_size, void* d_ws, size_t ws_size,
                              hipStream_t stream) {
    const float* x    = (const float*)d_in[0];   // [4096,768]
    const float* fc_w = (const float*)d_in[1];   // [2,1]
    const float* fc_b = (const float*)d_in[2];   // [2]
    float* out = (float*)d_out;                  // [4096,4096,2]

    unsigned short* xh = (unsigned short*)d_ws;
    unsigned short* xl = xh + (size_t)BDIM * DDIM;

    norm_split_kernel<<<BDIM, 256, 0, stream>>>(x, xh, xl);

    dim3 grid(BDIM / 64, BDIM / 64);
    gemm_bf16x3_kernel<<<grid, 128, 0, stream>>>(xh, xl, fc_w, fc_b, out);
}
